// DGSR_33698313404766
// MI455X (gfx1250) — matrix operand order repaired and checked by reading, not compiled, not run on hardware
//
#include <hip/hip_runtime.h>
#include <hip/hip_bf16.h>

// ---------------------------------------------------------------------------
// DGSR forward for MI455X (gfx1250, wave32).
// GEMMs run on v_wmma_f32_16x16x32_f16 (f16 in, f32 accumulate), fully
// unrolled over K via templates (K in {64,128,256}).
// Edge segment-softmax runs one wave32 per edge with float2 lanes,
// bit-flipped uint atomicMax for segment max, f32 atomic adds for sums.
// ---------------------------------------------------------------------------

typedef _Float16 v16h __attribute__((ext_vector_type(16)));
typedef float    v8f  __attribute__((ext_vector_type(8)));

#define DHID 64

// ----- helpers -------------------------------------------------------------

__device__ __forceinline__ unsigned flipF(float f) {
    unsigned u = __float_as_uint(f);
    return (u & 0x80000000u) ? ~u : (u | 0x80000000u);
}
__device__ __forceinline__ float unflipF(unsigned u) {
    return __uint_as_float((u & 0x80000000u) ? (u & 0x7FFFFFFFu) : ~u);
}

// ----- small utility kernels ----------------------------------------------

__global__ void k_cvt_f16(const float* __restrict__ s, _Float16* __restrict__ d, int n) {
    int i = blockIdx.x * 256 + threadIdx.x;
    if (i < n) d[i] = (_Float16)s[i];
}

__global__ void k_fill_f32(float* p, float v, int n) {
    int i = blockIdx.x * 256 + threadIdx.x;
    if (i < n) p[i] = v;
}

__global__ void k_fill_u32(unsigned* p, unsigned v, int n) {
    int i = blockIdx.x * 256 + threadIdx.x;
    if (i < n) p[i] = v;
}

// h[r, :] = emb[ids[r], :]   (n rows of 64)
__global__ void k_gather_rows(const float* __restrict__ emb, const int* __restrict__ ids,
                              float* __restrict__ out, int n) {
    int i = blockIdx.x * 256 + threadIdx.x;
    if (i < n * DHID) {
        int r = i >> 6, d = i & 63;
        out[i] = emb[(size_t)ids[r] * DHID + d];
    }
}

// gath[b, colOff + d] = h[idx[b], d]  with gath row stride 256
__global__ void k_gather_cat(const float* __restrict__ h, const int* __restrict__ idx,
                             float* __restrict__ gath, int Bn, int colOff) {
    int i = blockIdx.x * 256 + threadIdx.x;
    if (i < Bn * DHID) {
        int b = i >> 6, d = i & 63;
        gath[(size_t)b * 256 + colOff + d] = h[(size_t)idx[b] * DHID + d];
    }
}

// ----- edge kernels (one wave32 per edge, lane covers d = 2*lane, 2*lane+1) -

__global__ void k_edge_logits(const float* __restrict__ srcF, const float* __restrict__ dstF,
                              const int* __restrict__ src, const int* __restrict__ dst,
                              const float* __restrict__ tvec,
                              const float* __restrict__ freq, const float* __restrict__ phase,
                              float* __restrict__ eOut, unsigned* __restrict__ mOut, int En) {
    int e = blockIdx.x * 8 + (threadIdx.x >> 5);
    int lane = threadIdx.x & 31;
    if (e >= En) return;
    int s = src[e], t = dst[e];
    float tt = tvec[e];
    int d = lane * 2;
    float2 sv = *(const float2*)(srcF + (size_t)s * DHID + d);
    float2 dv = *(const float2*)(dstF + (size_t)t * DHID + d);
    float te0 = __cosf(tt * freq[d]     + phase[d]);
    float te1 = __cosf(tt * freq[d + 1] + phase[d + 1]);
    float acc = (te0 + sv.x) * dv.x + (te1 + sv.y) * dv.y;
#pragma unroll
    for (int off = 16; off > 0; off >>= 1) acc += __shfl_down(acc, off, 32);
    if (lane == 0) {
        float ev = acc * 0.125f;  // 1/sqrt(64)
        eOut[e] = ev;
        atomicMax(mOut + t, flipF(ev));
    }
}

__global__ void k_edge_expsum(const float* __restrict__ eBuf, const int* __restrict__ dst,
                              const unsigned* __restrict__ m, float* __restrict__ denom, int En) {
    int e = blockIdx.x * 256 + threadIdx.x;
    if (e >= En) return;
    int t = dst[e];
    atomicAdd(denom + t, __expf(eBuf[e] - unflipF(m[t])));
}

__global__ void k_edge_agg(const float* __restrict__ srcF,
                           const int* __restrict__ src, const int* __restrict__ dst,
                           const float* __restrict__ tvec,
                           const float* __restrict__ kfreq, const float* __restrict__ kphase,
                           const float* __restrict__ eBuf, const unsigned* __restrict__ m,
                           const float* __restrict__ denom, float* __restrict__ agg, int En) {
    int e = blockIdx.x * 8 + (threadIdx.x >> 5);
    int lane = threadIdx.x & 31;
    if (e >= En) return;
    int s = src[e], t = dst[e];
    float tt = tvec[e];
    float alpha = __expf(eBuf[e] - unflipF(m[t])) / denom[t];
    int d = lane * 2;
    float2 sv = *(const float2*)(srcF + (size_t)s * DHID + d);
    float k0 = __cosf(tt * kfreq[d]     + kphase[d])     + sv.x;
    float k1 = __cosf(tt * kfreq[d + 1] + kphase[d + 1]) + sv.y;
    atomicAdd(agg + (size_t)t * DHID + d,     alpha * k0);
    atomicAdd(agg + (size_t)t * DHID + d + 1, alpha * k1);
}

// ----- WMMA GEMM: C[M,64] = act( A[M,K] @ W[K,64] ) ------------------------
// Compile-time K = NKB*32. A is logically concat(A0[M,SPLIT_KB*32],
// A1[M,(NKB-SPLIT_KB)*32]) along K (SPLIT_KB==NKB -> A1 unused). W is f16
// row-major [K,64]. Block = 128 threads (4 waves); each wave owns a 16x16
// output tile; a block covers 16 rows x 64 cols. Full unroll over K-blocks
// gives straight-line v_wmma chains with loads overlapped.
// Fragment layouts per CDNA5 ISA 7.12.2 (wave32).

template <int NKB, int SPLIT_KB, int ACT>
__global__ __launch_bounds__(128)
void k_wmma_gemm(const float* __restrict__ A0, const float* __restrict__ A1,
                 const _Float16* __restrict__ W, float* __restrict__ C, int M) {
    constexpr int LD0 = SPLIT_KB * 32;
    constexpr int LD1 = (NKB - SPLIT_KB) * 32;
    int wave = threadIdx.x >> 5;           // 0..3 : 16-col block
    int lane = threadIdx.x & 31;
    int half = lane >> 4;                  // lane group 0 / 1
    int lr   = lane & 15;
    int rowBase = blockIdx.x * 16;
    int nb = wave * 16;
    if (rowBase >= M) return;
    int row = rowBase + lr;

    v8f c = {};
#pragma unroll
    for (int kb = 0; kb < NKB; ++kb) {
        const int kBase = kb * 32;
        v16h a, b;
#pragma unroll
        for (int v = 0; v < 8; ++v) {
            // A 16x32 f16 layout: lanes 0-15 hold K {0..7,16..23}, lanes 16-31
            // hold K {8..15,24..31}; VGPR v holds an even K pair.
            int kloc = ((v < 4) ? 2 * v : 2 * v + 8) + (half ? 8 : 0);
            int k = kBase + kloc;
            const float* ap = (kb < SPLIT_KB)                 // constant-folded
                ? (A0 + (size_t)row * LD0 + k)
                : (A1 + (size_t)row * LD1 + (k - LD0));
            float2 f = *(const float2*)ap;
            a[2 * v]     = (_Float16)f.x;
            a[2 * v + 1] = (_Float16)f.y;
            // B 32x16 f16 layout: same K order as A (lanes 0-15 hold K
            // {0..7,16..23}, lanes 16-31 hold K {8..15,24..31}); N = lane%16.
            int kw = kBase + ((v < 4) ? 2 * v : 2 * v + 8) + (half ? 8 : 0);
            b[2 * v]     = W[(size_t)kw * DHID + nb + lr];
            b[2 * v + 1] = W[(size_t)(kw + 1) * DHID + nb + lr];
        }
        c = __builtin_amdgcn_wmma_f32_16x16x32_f16(false, a, false, b,
                                                   (short)0, c, false, false);
    }
#pragma unroll
    for (int v = 0; v < 8; ++v) {
        int orow = rowBase + v + (half ? 8 : 0);
        float x = c[v];
        if (ACT == 1) x = (x > 0.f) ? x : (__expf(x) - 1.f);   // ELU
        C[(size_t)orow * DHID + nb + lr] = x;
    }
}

// ----- final dense head: out[b,j] = relu( U[b,:] @ Wd[:,j] + bd[j] ) -------

__global__ void k_dense(const float* __restrict__ U, const float* __restrict__ Wd,
                        const float* __restrict__ bd, float* __restrict__ out, int Bn) {
    int i = blockIdx.x * 256 + threadIdx.x;
    if (i >= Bn * 3) return;
    int b = i / 3, j = i % 3;
    float acc = bd[j];
#pragma unroll 8
    for (int d = 0; d < DHID; ++d) acc += U[(size_t)b * DHID + d] * Wd[d * 3 + j];
    out[i] = acc > 0.f ? acc : 0.f;
}

// ---------------------------------------------------------------------------

static inline int cdiv(int a, int b) { return (a + b - 1) / b; }

extern "C" void kernel_launch(void* const* d_in, const int* in_sizes, int n_in,
                              void* d_out, int out_size, void* d_ws, size_t ws_size,
                              hipStream_t stream) {
    // ---- inputs (setup_inputs dict order) ----
    const float* user_emb  = (const float*)d_in[0];
    const float* item_emb  = (const float*)d_in[1];
    const float* Wu        = (const float*)d_in[2];
    const float* Wi        = (const float*)d_in[3];
    const float* Wuu       = (const float*)d_in[4];
    const float* Wii       = (const float*)d_in[5];
    const float* u_freq    = (const float*)d_in[6];
    const float* u_phase   = (const float*)d_in[7];
    const float* uk_freq   = (const float*)d_in[8];
    const float* uk_phase  = (const float*)d_in[9];
    const float* i_freq    = (const float*)d_in[10];
    const float* i_phase   = (const float*)d_in[11];
    const float* ik_freq   = (const float*)d_in[12];
    const float* ik_phase  = (const float*)d_in[13];
    const float* unified_W = (const float*)d_in[14];
    const float* dense_W   = (const float*)d_in[15];
    const float* dense_b   = (const float*)d_in[16];
    const float* by_time   = (const float*)d_in[17];
    const float* pby_time  = (const float*)d_in[18];
    const int*   user_id   = (const int*)d_in[19];
    const int*   item_id   = (const int*)d_in[20];
    const int*   by_src    = (const int*)d_in[21];
    const int*   by_dst    = (const int*)d_in[22];
    const int*   pby_src   = (const int*)d_in[23];
    const int*   pby_dst   = (const int*)d_in[24];
    const int*   user_idx  = (const int*)d_in[25];
    const int*   last_item = (const int*)d_in[26];

    const int NUn = in_sizes[19];
    const int NIn = in_sizes[20];
    const int En  = in_sizes[21];
    const int Bn  = in_sizes[25];
    const int Ln  = in_sizes[2] / (DHID * DHID);
    const int CAT = (Ln + 1) * DHID;      // 256

    // ---- carve workspace ----
    char* p = (char*)d_ws;
    auto carve = [&](size_t bytes) -> void* {
        void* r = (void*)p;
        p += (bytes + 255) & ~(size_t)255;
        return r;
    };
    float* user_hA = (float*)carve((size_t)NUn * DHID * 4);
    float* user_hB = (float*)carve((size_t)NUn * DHID * 4);
    float* item_hA = (float*)carve((size_t)NIn * DHID * 4);
    float* item_hB = (float*)carve((size_t)NIn * DHID * 4);
    float* uh      = (float*)carve((size_t)NUn * DHID * 4);
    float* ih      = (float*)carve((size_t)NIn * DHID * 4);
    float* agg_u   = (float*)carve((size_t)NUn * DHID * 4);
    float* agg_i   = (float*)carve((size_t)NIn * DHID * 4);
    unsigned* m_u  = (unsigned*)carve((size_t)NUn * 4);
    unsigned* m_i  = (unsigned*)carve((size_t)NIn * 4);
    float* den_u   = (float*)carve((size_t)NUn * 4);
    float* den_i   = (float*)carve((size_t)NIn * 4);
    float* e_buf   = (float*)carve((size_t)En * 4);
    float* gath    = (float*)carve((size_t)Bn * CAT * 4);
    float* u64buf  = (float*)carve((size_t)Bn * DHID * 4);
    _Float16* WuF  = (_Float16*)carve((size_t)Ln * DHID * DHID * 2);
    _Float16* WiF  = (_Float16*)carve((size_t)Ln * DHID * DHID * 2);
    _Float16* WuuF = (_Float16*)carve((size_t)Ln * 2 * DHID * DHID * 2);
    _Float16* WiiF = (_Float16*)carve((size_t)Ln * 2 * DHID * DHID * 2);
    _Float16* WunF = (_Float16*)carve((size_t)CAT * DHID * 2);
    (void)ws_size; (void)n_in;

    // ---- weight conversion f32 -> f16 ----
    int nWu  = Ln * DHID * DHID;
    int nWuu = Ln * 2 * DHID * DHID;
    int nUni = CAT * DHID;
    k_cvt_f16<<<cdiv(nWu, 256), 256, 0, stream>>>(Wu, WuF, nWu);
    k_cvt_f16<<<cdiv(nWu, 256), 256, 0, stream>>>(Wi, WiF, nWu);
    k_cvt_f16<<<cdiv(nWuu, 256), 256, 0, stream>>>(Wuu, WuuF, nWuu);
    k_cvt_f16<<<cdiv(nWuu, 256), 256, 0, stream>>>(Wii, WiiF, nWuu);
    k_cvt_f16<<<cdiv(nUni, 256), 256, 0, stream>>>(unified_W, WunF, nUni);

    // ---- embedding gathers ----
    k_gather_rows<<<cdiv(NUn * DHID, 256), 256, 0, stream>>>(user_emb, user_id, user_hA, NUn);
    k_gather_rows<<<cdiv(NIn * DHID, 256), 256, 0, stream>>>(item_emb, item_id, item_hA, NIn);

    float* uOld = user_hA; float* uNew = user_hB;
    float* iOld = item_hA; float* iNew = item_hB;

    const dim3 blk256(256);
    const int gEdgeW = cdiv(En, 8);          // wave-per-edge kernels
    const int gEdgeT = cdiv(En, 256);        // thread-per-edge kernels

    for (int l = 0; l < Ln; ++l) {
        const _Float16* wu  = WuF  + (size_t)l * DHID * DHID;
        const _Float16* wi  = WiF  + (size_t)l * DHID * DHID;
        const _Float16* wuu = WuuF + (size_t)l * 2 * DHID * DHID;
        const _Float16* wii = WiiF + (size_t)l * 2 * DHID * DHID;
        const float* uf  = u_freq  + l * DHID; const float* up  = u_phase  + l * DHID;
        const float* ukf = uk_freq + l * DHID; const float* ukp = uk_phase + l * DHID;
        const float* itf = i_freq  + l * DHID; const float* itp = i_phase  + l * DHID;
        const float* ikf = ik_freq + l * DHID; const float* ikp = ik_phase + l * DHID;

        // uh = user_h @ Wu[l]; ih = item_h @ Wi[l]       (K=64, no act)
        k_wmma_gemm<2, 2, 0><<<NUn / 16, 128, 0, stream>>>(uOld, uOld, wu, uh, NUn);
        k_wmma_gemm<2, 2, 0><<<NIn / 16, 128, 0, stream>>>(iOld, iOld, wi, ih, NIn);

        // ---- 'by': item -> user ----
        k_fill_u32<<<cdiv(NUn, 256), blk256, 0, stream>>>(m_u, 0x007FFFFFu, NUn);  // flip(-inf)
        k_fill_f32<<<cdiv(NUn, 256), blk256, 0, stream>>>(den_u, 0.f, NUn);
        k_fill_f32<<<cdiv(NUn * DHID, 256), blk256, 0, stream>>>(agg_u, 0.f, NUn * DHID);
        k_edge_logits<<<gEdgeW, blk256, 0, stream>>>(ih, uh, by_src, by_dst, by_time,
                                                     uf, up, e_buf, m_u, En);
        k_edge_expsum<<<gEdgeT, blk256, 0, stream>>>(e_buf, by_dst, m_u, den_u, En);
        k_edge_agg<<<gEdgeW, blk256, 0, stream>>>(ih, by_src, by_dst, by_time, ukf, ukp,
                                                  e_buf, m_u, den_u, agg_u, En);

        // user update: elu(concat(agg_u, user_h) @ Wuu[l])   (K=128, split 64, ELU)
        k_wmma_gemm<4, 2, 1><<<NUn / 16, 128, 0, stream>>>(agg_u, uOld, wuu, uNew, NUn);

        // ---- 'pby': user -> item ----
        k_fill_u32<<<cdiv(NIn, 256), blk256, 0, stream>>>(m_i, 0x007FFFFFu, NIn);
        k_fill_f32<<<cdiv(NIn, 256), blk256, 0, stream>>>(den_i, 0.f, NIn);
        k_fill_f32<<<cdiv(NIn * DHID, 256), blk256, 0, stream>>>(agg_i, 0.f, NIn * DHID);
        k_edge_logits<<<gEdgeW, blk256, 0, stream>>>(uh, ih, pby_src, pby_dst, pby_time,
                                                     itf, itp, e_buf, m_i, En);
        k_edge_expsum<<<gEdgeT, blk256, 0, stream>>>(e_buf, pby_dst, m_i, den_i, En);
        k_edge_agg<<<gEdgeW, blk256, 0, stream>>>(uh, pby_src, pby_dst, pby_time, ikf, ikp,
                                                  e_buf, m_i, den_i, agg_i, En);

        // item update: elu(concat(agg_i, item_h) @ Wii[l])   (K=128, split 64, ELU)
        k_wmma_gemm<4, 2, 1><<<NIn / 16, 128, 0, stream>>>(agg_i, iOld, wii, iNew, NIn);

        // snapshot user layer l into the unified concat buffer
        k_gather_cat<<<cdiv(Bn * DHID, 256), blk256, 0, stream>>>(uNew, user_idx, gath,
                                                                  Bn, l * DHID);
        // swap
        float* t;
        t = uOld; uOld = uNew; uNew = t;
        t = iOld; iOld = iNew; iNew = t;
    }

    // last_item chunk
    k_gather_cat<<<cdiv(Bn * DHID, 256), blk256, 0, stream>>>(iOld, last_item, gath,
                                                              Bn, Ln * DHID);

    // unified = gath[B,256] @ unified_W[256,64]    (K=256, no act)
    k_wmma_gemm<8, 8, 0><<<Bn / 16, 128, 0, stream>>>(gath, gath, WunF, u64buf, Bn);

    // out = relu(unified @ dense_W + dense_b)
    k_dense<<<cdiv(Bn * 3, 256), blk256, 0, stream>>>(u64buf, dense_W, dense_b,
                                                      (float*)d_out, Bn);
}
